// GRUModel_25769803927
// MI455X (gfx1250) — compile-verified, matrix operand order checked by reading, not run on hardware
//
#include <hip/hip_runtime.h>

// ---------------------------------------------------------------------------
// Problem dims (fixed by the reference)
// ---------------------------------------------------------------------------
#define GB   64            // batch
#define GT   2048          // sequence length
#define GI   128           // input features
#define GH   256           // hidden
#define GC   64            // classes
#define G3H  (3 * GH)      // 768
#define GBT  (GB * GT)     // 131072 rows for the big GEMMs

typedef __attribute__((ext_vector_type(16))) _Float16 v16h;
typedef __attribute__((ext_vector_type(8)))  _Float16 v8h;
typedef __attribute__((ext_vector_type(8)))  float    v8f;

// ---------------------------------------------------------------------------
// CDNA5 async global->LDS copy (guarded: fall back to plain loads if the
// toolchain lacks the builtins).
// ---------------------------------------------------------------------------
#if defined(__has_builtin)
#if __has_builtin(__builtin_amdgcn_global_load_async_to_lds_b128) && \
    __has_builtin(__builtin_amdgcn_s_wait_asynccnt)
#define USE_ASYNC_LDS 1
#endif
#endif
#ifndef USE_ASYNC_LDS
#define USE_ASYNC_LDS 0
#endif

#if USE_ASYNC_LDS
// The builtin expects pointers to GCC-style int4 vectors:
//   param 1: global (AS1 / __device__) int4*   param 2: LDS (AS3) int4*
typedef int async_v4i __attribute__((vector_size(16)));
static __device__ __forceinline__ void async_copy16(const void* g, void* l) {
  // Generic->AS1 is numerically identity; a generic LDS pointer's low 32 bits
  // are the LDS byte offset (flat-aperture truncation rule, ISA 10.2).
  __builtin_amdgcn_global_load_async_to_lds_b128(
      (__attribute__((address_space(1))) async_v4i*)(uintptr_t)g,
      (__attribute__((address_space(3))) async_v4i*)(uint32_t)(uintptr_t)l,
      0, 0);
}
#endif

// ---------------------------------------------------------------------------
// WMMA fragment loader (16-bit A/B 16x32 layout, ISA 7.12.2):
//   lane 0-15  : row/col = lane,    K = 0..7  and 16..23
//   lane 16-31 : row/col = lane-16, K = 8..15 and 24..31
// ---------------------------------------------------------------------------
static __device__ __forceinline__ v16h frag_from_f16(const _Float16* __restrict__ base,
                                                     int ld, int row0, int k0, int lane) {
  const int n  = lane & 15;
  const int kb = (lane >> 4) << 3;
  const _Float16* p = base + (size_t)(row0 + n) * ld + (k0 + kb);
  v8h lo = *(const v8h*)(p);
  v8h hi = *(const v8h*)(p + 16);
  v16h f;
#pragma unroll
  for (int j = 0; j < 8; ++j) { f[j] = lo[j]; f[8 + j] = hi[j]; }
  return f;
}

// ---------------------------------------------------------------------------
// Generic WMMA GEMM:  Out[m,n] = sum_k A[m,k] * W[n,k] + bias[n]
//   A: f16 row-major [M,K];  W: f16 row-major [N,K] (so B = W^T)
//   If Tperm>0: output row m=(b*Tperm+t) is stored at row (t*Bperm+b)
//   Block tile 128x64, 256 threads = 8 waves in 4x2, each wave 32x32 (4 tiles)
// ---------------------------------------------------------------------------
template <typename OutT>
__global__ __launch_bounds__(256) void wmma_gemm_bias(
    const _Float16* __restrict__ A, const _Float16* __restrict__ W,
    const float* __restrict__ bias, OutT* __restrict__ Out,
    int N, int K, int Tperm, int Bperm) {
  const int tid   = threadIdx.x;
  const int lane  = tid & 31;
  const int wave  = tid >> 5;
  const int waveM = wave & 3;
  const int waveN = wave >> 2;
  const int rowbase = blockIdx.x * 128 + waveM * 32;
  const int colbase = blockIdx.y * 64  + waveN * 32;

  v8f acc[2][2];
#pragma unroll
  for (int i = 0; i < 2; ++i)
#pragma unroll
    for (int j = 0; j < 2; ++j) acc[i][j] = {};

#pragma unroll 2
  for (int k0 = 0; k0 < K; k0 += 32) {
    v16h a0 = frag_from_f16(A, K, rowbase,      k0, lane);
    v16h a1 = frag_from_f16(A, K, rowbase + 16, k0, lane);
    v16h b0 = frag_from_f16(W, K, colbase,      k0, lane);
    v16h b1 = frag_from_f16(W, K, colbase + 16, k0, lane);
    acc[0][0] = __builtin_amdgcn_wmma_f32_16x16x32_f16(false, a0, false, b0, (short)0, acc[0][0], false, false);
    acc[0][1] = __builtin_amdgcn_wmma_f32_16x16x32_f16(false, a0, false, b1, (short)0, acc[0][1], false, false);
    acc[1][0] = __builtin_amdgcn_wmma_f32_16x16x32_f16(false, a1, false, b0, (short)0, acc[1][0], false, false);
    acc[1][1] = __builtin_amdgcn_wmma_f32_16x16x32_f16(false, a1, false, b1, (short)0, acc[1][1], false, false);
  }

  // C/D layout: lanes 0-15 -> M=r, lanes 16-31 -> M=8+r; N = lane&15
  const int mo = (lane >> 4) << 3;
  const int nn = lane & 15;
#pragma unroll
  for (int i = 0; i < 2; ++i) {
#pragma unroll
    for (int j = 0; j < 2; ++j) {
      const int col = colbase + j * 16 + nn;
      const float bv = bias[col];
#pragma unroll
      for (int r = 0; r < 8; ++r) {
        const int m = rowbase + i * 16 + mo + r;
        int orow;
        if (Tperm > 0) {
          const int bb = m / Tperm;
          const int tt = m - bb * Tperm;
          orow = tt * Bperm + bb;
        } else {
          orow = m;
        }
        Out[(size_t)orow * N + col] = (OutT)(acc[i][j][r] + bv);
      }
    }
  }
}

// ---------------------------------------------------------------------------
// Persistent single-workgroup GRU recurrence over T steps.
//   gx  : [T, B, 3H] f16 precomputed input-side gates (b_ih already added)
//   Whh : [3H, H] f16 (streams from L2 each step; 393KB, L2-resident)
//   bhh : [3H] f32  (cached in LDS)
//   hout: [B, T, H] f16 layer output (consumed only as WMMA A-operand later)
//
// LDS (320KB WGP budget): h f16 32KB | gh f16 96KB | gx-stage f16 96KB |
// bias f32 3KB  ->  ~227KB.  512 threads = 16 waves; wave w owns N-tiles
// 3w..3w+2 x 4 M-tiles = 12 WMMA accumulators -> full [64,768] gate GEMM/step.
// Step t's gx slice is async-copied to LDS while the WMMAs run.
// ---------------------------------------------------------------------------
__global__ __launch_bounds__(512) void gru_recurrent(
    const _Float16* __restrict__ gx, const _Float16* __restrict__ Whh,
    const float* __restrict__ bhh, _Float16* __restrict__ hout, int T) {
  extern __shared__ char smem[];
  _Float16* hsh = (_Float16*)smem;            // [64][256]
  _Float16* ghs = hsh + GB * GH;              // [64][768]
  _Float16* gxs = ghs + GB * G3H;             // [64][768] async stage
  float*    bsh = (float*)(gxs + GB * G3H);   // [768]

  const int tid  = threadIdx.x;
  const int lane = tid & 31;
  const int wave = tid >> 5;  // 0..15

  for (int i = tid; i < GB * GH; i += 512) hsh[i] = (_Float16)0.0f;
  if (tid < G3H) bsh[tid] = bhh[tid];
  __syncthreads();

  for (int t = 0; t < T; ++t) {
    const _Float16* gxt = gx + (size_t)t * (GB * G3H);

#if USE_ASYNC_LDS
    // Kick off async fill of this step's gx slice (6144 x 16B chunks);
    // it completes in the shadow of the 96 WMMAs below.
#pragma unroll
    for (int j = 0; j < 12; ++j) {
      const int chunk = tid + j * 512;
      async_copy16(gxt + chunk * 8, gxs + chunk * 8);
    }
#else
    if (t + 1 < T) {
      const _Float16* nx = gxt + GB * G3H;
#pragma unroll
      for (int j = 0; j < 3; ++j)
        __builtin_prefetch(nx + tid * 96 + j * 32, 0, 1);
    }
#endif

    // ---- gh = h @ Whh^T  (M=64, N=768, K=256) ----
    v8f acc[4][3];
#pragma unroll
    for (int mt = 0; mt < 4; ++mt)
#pragma unroll
      for (int bt = 0; bt < 3; ++bt) acc[mt][bt] = {};

#pragma unroll 2
    for (int kk = 0; kk < 8; ++kk) {
      const int k0 = kk * 32;
      v16h a[4];
#pragma unroll
      for (int mt = 0; mt < 4; ++mt)
        a[mt] = frag_from_f16(hsh, GH, mt * 16, k0, lane);
      v16h b[3];
#pragma unroll
      for (int bt = 0; bt < 3; ++bt)
        b[bt] = frag_from_f16(Whh, GH, (wave * 3 + bt) * 16, k0, lane);
#pragma unroll
      for (int mt = 0; mt < 4; ++mt)
#pragma unroll
        for (int bt = 0; bt < 3; ++bt)
          acc[mt][bt] = __builtin_amdgcn_wmma_f32_16x16x32_f16(
              false, a[mt], false, b[bt], (short)0, acc[mt][bt], false, false);
    }

    // Spill gate pre-activations to LDS (f16).
    const int nn = lane & 15;
    const int mo = (lane >> 4) << 3;
#pragma unroll
    for (int mt = 0; mt < 4; ++mt)
#pragma unroll
      for (int bt = 0; bt < 3; ++bt) {
        const int col = (wave * 3 + bt) * 16 + nn;
#pragma unroll
        for (int r = 0; r < 8; ++r)
          ghs[(mt * 16 + mo + r) * G3H + col] = (_Float16)acc[mt][bt][r];
      }

#if USE_ASYNC_LDS
    __builtin_amdgcn_s_wait_asynccnt(0);  // my async chunks are in LDS
#endif
    __syncthreads();                      // everyone's chunks + gh spill visible

    // ---- fused GRU cell update: 64*256 elems, 32 per thread ----
#if USE_ASYNC_LDS
    const _Float16* __restrict__ gsrc = gxs;
#else
    const _Float16* __restrict__ gsrc = gxt;
#endif
#pragma unroll 4
    for (int i = 0; i < 32; ++i) {
      const int idx = tid + i * 512;
      const int b = idx >> 8;        // /256
      const int c = idx & 255;       // %256
      const float xr = (float)gsrc[b * G3H + c];
      const float xz = (float)gsrc[b * G3H + GH + c];
      const float xn = (float)gsrc[b * G3H + 2 * GH + c];
      const float hr = (float)ghs[b * G3H + c] + bsh[c];
      const float hz = (float)ghs[b * G3H + GH + c] + bsh[GH + c];
      const float hn = (float)ghs[b * G3H + 2 * GH + c] + bsh[2 * GH + c];
      const float rg = 1.0f / (1.0f + __expf(-(xr + hr)));
      const float zg = 1.0f / (1.0f + __expf(-(xz + hz)));
      const float nv = tanhf(xn + rg * hn);
      const float hp = (float)hsh[b * GH + c];
      const float hnew = (1.0f - zg) * nv + zg * hp;
      hsh[b * GH + c] = (_Float16)hnew;
      hout[(size_t)b * T * GH + (size_t)t * GH + c] = (_Float16)hnew;
    }
    __syncthreads();
  }
}

// ---------------------------------------------------------------------------
__global__ void cvt_f32_f16(const float* __restrict__ s, _Float16* __restrict__ d, int n) {
  const int i = blockIdx.x * 256 + threadIdx.x;
  if (i < n) d[i] = (_Float16)s[i];
}

// ---------------------------------------------------------------------------
extern "C" void kernel_launch(void* const* d_in, const int* in_sizes, int n_in,
                              void* d_out, int out_size, void* d_ws, size_t ws_size,
                              hipStream_t stream) {
  (void)in_sizes; (void)n_in; (void)out_size; (void)ws_size;

  const float* x    = (const float*)d_in[0];
  const float* Wih0 = (const float*)d_in[1];
  const float* Whh0 = (const float*)d_in[2];
  const float* bih0 = (const float*)d_in[3];
  const float* bhh0 = (const float*)d_in[4];
  const float* Wih1 = (const float*)d_in[5];
  const float* Whh1 = (const float*)d_in[6];
  const float* bih1 = (const float*)d_in[7];
  const float* bhh1 = (const float*)d_in[8];
  const float* fcw  = (const float*)d_in[9];
  const float* fcb  = (const float*)d_in[10];
  float* out = (float*)d_out;

  // Workspace carve-up.
  char* ws = (char*)d_ws;
  size_t off = 0;
  auto take = [&](size_t bytes) { char* p = ws + off; off += (bytes + 255) & ~(size_t)255; return p; };
  _Float16* xh    = (_Float16*)take((size_t)GBT * GI * sizeof(_Float16)); // [B*T, I]
  _Float16* Wih0h = (_Float16*)take((size_t)G3H * GI * sizeof(_Float16));
  _Float16* Whh0h = (_Float16*)take((size_t)G3H * GH * sizeof(_Float16));
  _Float16* Wih1h = (_Float16*)take((size_t)G3H * GH * sizeof(_Float16));
  _Float16* Whh1h = (_Float16*)take((size_t)G3H * GH * sizeof(_Float16));
  _Float16* fcwh  = (_Float16*)take((size_t)GC  * GH * sizeof(_Float16));
  _Float16* gxbuf = (_Float16*)take((size_t)GBT * G3H * sizeof(_Float16)); // [T,B,3H]
  _Float16* h0buf = (_Float16*)take((size_t)GBT * GH  * sizeof(_Float16)); // [B,T,H]
  _Float16* h1buf = (_Float16*)take((size_t)GBT * GH  * sizeof(_Float16)); // [B,T,H]

  // 1) one-time f32 -> f16 conversions (x + all weights)
  cvt_f32_f16<<<(GBT * GI + 255) / 256, 256, 0, stream>>>(x, xh, GBT * GI);
  cvt_f32_f16<<<(G3H * GI + 255) / 256, 256, 0, stream>>>(Wih0, Wih0h, G3H * GI);
  cvt_f32_f16<<<(G3H * GH + 255) / 256, 256, 0, stream>>>(Whh0, Whh0h, G3H * GH);
  cvt_f32_f16<<<(G3H * GH + 255) / 256, 256, 0, stream>>>(Wih1, Wih1h, G3H * GH);
  cvt_f32_f16<<<(G3H * GH + 255) / 256, 256, 0, stream>>>(Whh1, Whh1h, G3H * GH);
  cvt_f32_f16<<<(GC  * GH + 255) / 256, 256, 0, stream>>>(fcw,  fcwh,  GC  * GH);

  const size_t smem = (size_t)(GB * GH + 2 * GB * G3H) * sizeof(_Float16)
                    + (size_t)G3H * sizeof(float);   // 232448 B
  (void)hipFuncSetAttribute((const void*)gru_recurrent,
                            hipFuncAttributeMaxDynamicSharedMemorySize, (int)smem);

  // 2) layer-0 input-side gates: [131072,128] x [128,768] -> [T,B,3H] f16
  dim3 gGates(GBT / 128, G3H / 64);
  wmma_gemm_bias<_Float16><<<gGates, 256, 0, stream>>>(xh, Wih0h, bih0, gxbuf, G3H, GI, GT, GB);

  // 3) layer-0 recurrence (persistent workgroup, h in LDS)
  gru_recurrent<<<1, 512, smem, stream>>>(gxbuf, Whh0h, bhh0, h0buf, GT);

  // 4) layer-1 input-side gates: [131072,256] x [256,768] -> [T,B,3H] f16
  wmma_gemm_bias<_Float16><<<gGates, 256, 0, stream>>>(h0buf, Wih1h, bih1, gxbuf, G3H, GH, GT, GB);

  // 5) layer-1 recurrence
  gru_recurrent<<<1, 512, smem, stream>>>(gxbuf, Whh1h, bhh1, h1buf, GT);

  // 6) head: [131072,256] x [256,64] -> d_out [B,T,C] f32
  dim3 gHead(GBT / 128, GC / 64);
  wmma_gemm_bias<float><<<gHead, 256, 0, stream>>>(h1buf, fcwh, fcb, out, GC, GH, 0, 0);
}
